// BinarizedNeuralNetwork_78108275245423
// MI455X (gfx1250) — 8-bit matrix operand order repaired and checked by reading, compiled, not run on hardware
//
#include <hip/hip_runtime.h>
#include <hip/hip_bf16.h>
#include <stdint.h>

typedef __attribute__((ext_vector_type(8))) int v8i;

#define BNN_EPS 1e-5f

__global__ void binarize_weights_kernel(const float* __restrict__ w,
                                        int8_t* __restrict__ dst,
                                        int OC, int IC, int Kp) {
    int idx = blockIdx.x * blockDim.x + threadIdx.x;
    if (idx >= OC * Kp) return;
    int oc = idx / Kp, k = idx % Kp;
    int8_t s = 0;
    if (k < 9 * IC) {
        int tap = k / IC, ic = k % IC;
        float v = w[oc * IC * 9 + ic * 9 + tap];
        s = (int8_t)((v > 0.f) - (v < 0.f));
    }
    dst[idx] = s;
}

__global__ void conv1_stats_kernel(const float* __restrict__ x,
                                   const float* __restrict__ w1,
                                   const float* __restrict__ b1,
                                   float* __restrict__ gsum,
                                   float* __restrict__ gsq) {
    __shared__ float sw[288];
    __shared__ float sb[32];
    for (int i = threadIdx.x; i < 288; i += blockDim.x) sw[i] = w1[i];
    for (int i = threadIdx.x; i < 32; i += blockDim.x) sb[i] = b1[i];
    __syncthreads();

    float psum[32], psq[32];
#pragma unroll
    for (int c = 0; c < 32; ++c) { psum[c] = 0.f; psq[c] = 0.f; }

    const int P = 128 * 128 * 128;
    int stride = gridDim.x * blockDim.x;
    for (int p = blockIdx.x * blockDim.x + threadIdx.x; p < P; p += stride) {
        int n = p >> 14;
        int h = (p >> 7) & 127;
        int w = p & 127;
        const float* xn = x + (n << 14);
        float xv[9];
#pragma unroll
        for (int t = 0; t < 9; ++t) {
            int dy = t / 3, dx = t % 3;
            int hh = h + dy - 1, ww = w + dx - 1;
            bool inb = (hh >= 0) && (hh < 128) && (ww >= 0) && (ww < 128);
            xv[t] = inb ? xn[(hh << 7) + ww] : 0.f;
        }
#pragma unroll
        for (int c = 0; c < 32; ++c) {
            float acc = sb[c];
#pragma unroll
            for (int t = 0; t < 9; ++t) acc = fmaf(xv[t], sw[c * 9 + t], acc);
            psum[c] += acc;
            psq[c] += acc * acc;
        }
    }

    int lane = threadIdx.x & 31;
#pragma unroll
    for (int c = 0; c < 32; ++c) {
        float s = psum[c], q = psq[c];
        for (int m = 16; m >= 1; m >>= 1) {
            s += __shfl_xor(s, m);
            q += __shfl_xor(q, m);
        }
        if (lane == 0) {
            atomicAdd(&gsum[c], s);
            atomicAdd(&gsq[c], q);
        }
    }
}

__global__ void conv1_binarize_kernel(const float* __restrict__ x,
                                      const float* __restrict__ w1,
                                      const float* __restrict__ b1,
                                      const float* __restrict__ scl,
                                      const float* __restrict__ shf,
                                      int8_t* __restrict__ s1) {
    __shared__ float sw[288];
    __shared__ float sb[32], ssc[32], ssh[32];
    for (int i = threadIdx.x; i < 288; i += blockDim.x) sw[i] = w1[i];
    if (threadIdx.x < 32) {
        sb[threadIdx.x]  = b1[threadIdx.x];
        ssc[threadIdx.x] = scl[threadIdx.x];
        ssh[threadIdx.x] = shf[threadIdx.x];
    }
    __syncthreads();

    int p = blockIdx.x * blockDim.x + threadIdx.x;
    int n = p >> 14;
    int h = (p >> 7) & 127;
    int w = p & 127;
    const float* xn = x + (n << 14);
    float xv[9];
#pragma unroll
    for (int t = 0; t < 9; ++t) {
        int dy = t / 3, dx = t % 3;
        int hh = h + dy - 1, ww = w + dx - 1;
        bool inb = (hh >= 0) && (hh < 128) && (ww >= 0) && (ww < 128);
        xv[t] = inb ? xn[(hh << 7) + ww] : 0.f;
    }
    int* outw = (int*)s1 + p * 8;
#pragma unroll
    for (int g = 0; g < 8; ++g) {
        int packed = 0;
#pragma unroll
        for (int u = 0; u < 4; ++u) {
            int c = g * 4 + u;
            float acc = sb[c];
#pragma unroll
            for (int t = 0; t < 9; ++t) acc = fmaf(xv[t], sw[c * 9 + t], acc);
            float y = acc * ssc[c] + ssh[c];
            int s = (y > 0.f) - (y < 0.f);
            packed |= (s & 0xFF) << (8 * u);
        }
        outw[g] = packed;
    }
}

template <int H, int W, int Cout>
__global__ void binconv_pool_kernel(const int8_t* __restrict__ in,
                                    const int8_t* __restrict__ wb,
                                    const float* __restrict__ alphaPtr,
                                    float* __restrict__ out) {
    const int Kp = 320;
    __shared__ __align__(16) int8_t tile[4][40][32];

    const int WG = W >> 5;
    int tmp = blockIdx.x;
    int wg = tmp % WG; tmp /= WG;
    int hp = tmp % (H >> 1); tmp /= (H >> 1);
    int n = tmp;
    int h0 = hp << 1;
    int wbase = wg << 5;

    for (int d = threadIdx.x; d < 1280; d += blockDim.x) {
        int r = d / 320;
        int rem = d - r * 320;
        int j = rem >> 3;
        int icg = (rem & 7) << 2;
        int hh = h0 - 1 + r;
        int ww = wbase - 1 + j;
        int val = 0;
        if (j < 34 && hh >= 0 && hh < H && ww >= 0 && ww < W)
            val = *(const int*)(in + ((n * H + hh) * W + ww) * 32 + icg);
        *(int*)&tile[r][j][icg] = val;
    }
    __syncthreads();

    float alpha = alphaPtr[0];
    int lane = threadIdx.x & 31;
    int wid = threadIdx.x >> 5;
    int sub = wid & 1;
    int ocb = (wid >> 1) << 4;

    int mrow = lane & 15;
    int half = lane >> 4;
    int colb = (sub << 4) + mrow;
    const int8_t* wrow = wb + (ocb + mrow) * Kp + (half << 4);

    v8i acc0 = {0, 0, 0, 0, 0, 0, 0, 0};
    v8i acc1 = {0, 0, 0, 0, 0, 0, 0, 0};

#pragma unroll
    for (int ck = 0; ck < 5; ++ck) {
        const int kk = ck * 64;
        v8i a;
#pragma unroll
        for (int v = 0; v < 8; ++v) {
            const int kofs = ((v >> 2) << 5) + ((v & 3) << 2);
            a[v] = *(const int*)(wrow + kk + kofs);
        }
        v8i b0, b1;
#pragma unroll
        for (int v = 0; v < 8; ++v) {
            const int klocal = ((v >> 2) << 5) + ((v & 3) << 2);
            const int kgA = kk + klocal;
            const int kgB = kk + klocal + 16;
            int kg = half ? kgB : kgA;
            const int tapA = kgA >> 5, tapB = kgB >> 5;
            int tap = half ? tapB : tapA;
            int icg = kg & 31;
            int dy = tap / 3, dx = tap - dy * 3;
            int v0 = 0, v1 = 0;
            if (kg < 288) {
                v0 = *(const int*)&tile[dy + 0][colb + dx][icg];
                v1 = *(const int*)&tile[dy + 1][colb + dx][icg];
            }
            b0[v] = v0;
            b1[v] = v1;
        }
        acc0 = __builtin_amdgcn_wmma_i32_16x16x64_iu8(true, a, true, b0, acc0, false, false);
        acc1 = __builtin_amdgcn_wmma_i32_16x16x64_iu8(true, a, true, b1, acc1, false, false);
    }

    const int Hp = H >> 1, Wp = W >> 1;
#pragma unroll
    for (int r = 0; r < 8; ++r) {
        float f0 = alpha * (float)acc0[r];
        float f1 = alpha * (float)acc1[r];
        float m = fmaxf(f0, f1);
        float o = __shfl_xor(m, 1);
        m = fmaxf(m, o);
        if ((lane & 1) == 0) {
            int oc = ocb + r + (half << 3);
            int wp = (wbase + colb) >> 1;
            out[((n * Hp + hp) * Wp + wp) * Cout + oc] = m;
        }
    }
}

__global__ void binconv4_gap_kernel(const int8_t* __restrict__ in,
                                    const int8_t* __restrict__ wb,
                                    const float* __restrict__ alphaPtr,
                                    float* __restrict__ sums4) {
    const int H = 32, W = 32, Kp = 576;
    __shared__ __align__(16) int8_t tile[3][40][64];

    int h = blockIdx.x & 31;
    int n = blockIdx.x >> 5;

    for (int d = threadIdx.x; d < 1920; d += blockDim.x) {
        int r = d / 640;
        int rem = d - r * 640;
        int j = rem >> 4;
        int icg = (rem & 15) << 2;
        int hh = h - 1 + r;
        int ww = j - 1;
        int val = 0;
        if (j < 34 && hh >= 0 && hh < H && ww >= 0 && ww < W)
            val = *(const int*)(in + ((n * H + hh) * W + ww) * 64 + icg);
        *(int*)&tile[r][j][icg] = val;
    }
    __syncthreads();

    float alpha = alphaPtr[0];
    int lane = threadIdx.x & 31;
    int wid = threadIdx.x >> 5;
    int sub = wid & 1;
    int ocb = (wid >> 1) << 4;

    int mrow = lane & 15;
    int half = lane >> 4;
    int colb = (sub << 4) + mrow;
    const int8_t* wrow = wb + (ocb + mrow) * Kp + (half << 4);

    v8i acc = {0, 0, 0, 0, 0, 0, 0, 0};
#pragma unroll
    for (int tap = 0; tap < 9; ++tap) {
        const int kk = tap << 6;
        v8i a;
#pragma unroll
        for (int v = 0; v < 8; ++v) {
            const int kofs = ((v >> 2) << 5) + ((v & 3) << 2);
            a[v] = *(const int*)(wrow + kk + kofs);
        }
        const int dy = tap / 3, dx = tap - dy * 3;
        v8i b;
#pragma unroll
        for (int v = 0; v < 8; ++v) {
            const int klocal = ((v >> 2) << 5) + ((v & 3) << 2);
            int icg = klocal + (half << 4);
            b[v] = *(const int*)&tile[dy][colb + dx][icg];
        }
        acc = __builtin_amdgcn_wmma_i32_16x16x64_iu8(true, a, true, b, acc, false, false);
    }

#pragma unroll
    for (int r = 0; r < 8; ++r) {
        float f = alpha * (float)acc[r];
        f += __shfl_xor(f, 1);
        f += __shfl_xor(f, 2);
        f += __shfl_xor(f, 4);
        f += __shfl_xor(f, 8);
        if (mrow == 0)
            atomicAdd(&sums4[n * 128 + ocb + r + (half << 3)], f);
    }
}

__global__ void bn_stats_kernel(const float* __restrict__ buf, long L, int C,
                                float* __restrict__ gsum, float* __restrict__ gsq) {
    __shared__ float ls[64], lq[64];
    if ((int)threadIdx.x < C) { ls[threadIdx.x] = 0.f; lq[threadIdx.x] = 0.f; }
    __syncthreads();
    long stride = (long)gridDim.x * blockDim.x;
    for (long i = blockIdx.x * (long)blockDim.x + threadIdx.x; i < L; i += stride) {
        float v = buf[i];
        int c = (int)(i & (long)(C - 1));
        atomicAdd(&ls[c], v);
        atomicAdd(&lq[c], v * v);
    }
    __syncthreads();
    if ((int)threadIdx.x < C) {
        atomicAdd(&gsum[threadIdx.x], ls[threadIdx.x]);
        atomicAdd(&gsq[threadIdx.x],  lq[threadIdx.x]);
    }
}

__global__ void finalize_stats_kernel(const float* __restrict__ gsum,
                                      const float* __restrict__ gsq,
                                      const float* __restrict__ gamma,
                                      const float* __restrict__ beta,
                                      float invCount, int C,
                                      float* __restrict__ scl,
                                      float* __restrict__ shf) {
    int c = blockIdx.x * blockDim.x + threadIdx.x;
    if (c < C) {
        float mu  = gsum[c] * invCount;
        float var = gsq[c] * invCount - mu * mu;
        float sc  = gamma[c] * rsqrtf(var + BNN_EPS);
        scl[c] = sc;
        shf[c] = beta[c] - mu * sc;
    }
}

__global__ void binarize_kernel(const float* __restrict__ buf, long L, int C,
                                const float* __restrict__ scl,
                                const float* __restrict__ shf,
                                int8_t* __restrict__ out) {
    long i = blockIdx.x * (long)blockDim.x + threadIdx.x;
    if (i < L) {
        int c = (int)(i & (long)(C - 1));
        float y = buf[i] * scl[c] + shf[c];
        out[i] = (int8_t)((y > 0.f) - (y < 0.f));
    }
}

__global__ void zero_kernel(float* __restrict__ p, long L) {
    long i = blockIdx.x * (long)blockDim.x + threadIdx.x;
    if (i < L) p[i] = 0.f;
}

__global__ void fc_kernel(const float* __restrict__ sums4,
                          const float* __restrict__ wf,
                          const float* __restrict__ bf,
                          float* __restrict__ out) {
    int t = blockIdx.x * blockDim.x + threadIdx.x;
    if (t >= 128 * 10) return;
    int n = t / 10, j = t % 10;
    float acc = bf[j];
    const float* s  = sums4 + n * 128;
    const float* wr = wf + j * 128;
#pragma unroll 8
    for (int c = 0; c < 128; ++c)
        acc = fmaf(s[c] * (1.f / 1024.f), wr[c], acc);
    out[t] = acc;
}

extern "C" void kernel_launch(void* const* d_in, const int* in_sizes, int n_in,
                              void* d_out, int out_size, void* d_ws, size_t ws_size,
                              hipStream_t stream) {
    (void)in_sizes; (void)n_in; (void)out_size; (void)ws_size;
    const float* x   = (const float*)d_in[0];
    const float* w1  = (const float*)d_in[1];
    const float* b1  = (const float*)d_in[2];
    const float* g1  = (const float*)d_in[3];
    const float* be1 = (const float*)d_in[4];
    const float* w2  = (const float*)d_in[5];
    const float* a2  = (const float*)d_in[6];
    const float* g2  = (const float*)d_in[7];
    const float* be2 = (const float*)d_in[8];
    const float* w3  = (const float*)d_in[9];
    const float* a3  = (const float*)d_in[10];
    const float* g3  = (const float*)d_in[11];
    const float* be3 = (const float*)d_in[12];
    const float* w4  = (const float*)d_in[13];
    const float* a4  = (const float*)d_in[14];
    const float* wf  = (const float*)d_in[15];
    const float* bf  = (const float*)d_in[16];
    float* out = (float*)d_out;

    char* ws = (char*)d_ws;
    int8_t* wb2   = (int8_t*)(ws + 0);
    int8_t* wb3   = (int8_t*)(ws + 10240);
    int8_t* wb4   = (int8_t*)(ws + 30720);
    float*  gsum  = (float*)(ws + 104448);
    float*  gsq   = gsum + 64;
    float*  scl   = (float*)(ws + 104960);
    float*  shf   = scl + 64;
    int8_t* s1    = (int8_t*)(ws + 105472);
    float*  p3    = (float*)(ws + 105472);
    float*  p2    = (float*)(ws + 67214336);
    int8_t* b3    = (int8_t*)(ws + 67214336);
    int8_t* b2    = (int8_t*)(ws + 134323200);
    float*  sums4 = (float*)(ws + 151100416);

    binarize_weights_kernel<<<(32 * 320 + 255) / 256, 256, 0, stream>>>(w2, wb2, 32, 32, 320);
    binarize_weights_kernel<<<(64 * 320 + 255) / 256, 256, 0, stream>>>(w3, wb3, 64, 32, 320);
    binarize_weights_kernel<<<(128 * 576 + 255) / 256, 256, 0, stream>>>(w4, wb4, 128, 64, 576);

    zero_kernel<<<1, 128, 0, stream>>>(gsum, 128);
    conv1_stats_kernel<<<512, 256, 0, stream>>>(x, w1, b1, gsum, gsq);
    finalize_stats_kernel<<<1, 32, 0, stream>>>(gsum, gsq, g1, be1, 1.f / 2097152.f, 32, scl, shf);
    conv1_binarize_kernel<<<2097152 / 256, 256, 0, stream>>>(x, w1, b1, scl, shf, s1);

    binconv_pool_kernel<128, 128, 32><<<128 * 64 * 4, 128, 0, stream>>>(s1, wb2, a2, p2);
    zero_kernel<<<1, 128, 0, stream>>>(gsum, 128);
    bn_stats_kernel<<<2048, 256, 0, stream>>>(p2, 16777216L, 32, gsum, gsq);
    finalize_stats_kernel<<<1, 32, 0, stream>>>(gsum, gsq, g2, be2, 1.f / 524288.f, 32, scl, shf);
    binarize_kernel<<<16777216 / 256, 256, 0, stream>>>(p2, 16777216L, 32, scl, shf, b2);

    binconv_pool_kernel<64, 64, 64><<<128 * 32 * 2, 256, 0, stream>>>(b2, wb3, a3, p3);
    zero_kernel<<<1, 128, 0, stream>>>(gsum, 128);
    bn_stats_kernel<<<2048, 256, 0, stream>>>(p3, 8388608L, 64, gsum, gsq);
    finalize_stats_kernel<<<1, 64, 0, stream>>>(gsum, gsq, g3, be3, 1.f / 131072.f, 64, scl, shf);
    binarize_kernel<<<8388608 / 256, 256, 0, stream>>>(p3, 8388608L, 64, scl, shf, b3);

    zero_kernel<<<(16384 + 255) / 256, 256, 0, stream>>>(sums4, 16384);
    binconv4_gap_kernel<<<128 * 32, 512, 0, stream>>>(b3, wb4, a4, sums4);

    fc_kernel<<<(1280 + 255) / 256, 256, 0, stream>>>(sums4, wf, bf, out);
}
